// MultiheadAtt_62294205661616
// MI455X (gfx1250) — matrix operand order repaired and checked by reading, not compiled, not run on hardware
//
#include <hip/hip_runtime.h>

// ---------------- problem constants ----------------
constexpr int kB   = 2;
constexpr int kS   = 2048;
constexpr int kH   = 1024;     // hidden
constexpr int kNH  = 16;       // heads
constexpr int kHD  = 64;       // head dim
constexpr int kM   = kB * kS;  // 4096 rows
constexpr float kScale = 0.125f;   // 1/sqrt(64)
constexpr float kLnEps = 1e-5f;

// ---------------- CDNA5 WMMA types ----------------
typedef __attribute__((ext_vector_type(16))) __bf16          v16bf;
typedef __attribute__((ext_vector_type(8)))  float           v8f;
typedef __attribute__((ext_vector_type(4)))  float           f32x4;
typedef __attribute__((ext_vector_type(4)))  unsigned int    u32x4;
typedef __attribute__((ext_vector_type(4)))  unsigned short  u16x4;

struct frag_bits { u32x4 lo, hi; };   // 32 bytes == v16bf

__device__ inline unsigned short f2bf_bits(float f) {        // RNE float -> bf16 bits
  unsigned u = __builtin_bit_cast(unsigned, f);
  u += 0x7fffu + ((u >> 16) & 1u);
  return (unsigned short)(u >> 16);
}

__device__ inline v8f wmma_bf16(v16bf a, v16bf b, v8f c) {
  return __builtin_amdgcn_wmma_f32_16x16x32_bf16(false, a, false, b, (short)0, c, false, false);
}

// ---- fragment loaders (wave32, contiguous k order), 128-bit vector loads ----
// A 16x32 bf16: lane m=lane&15, hf=lane>>4; element (2v+e) holds k = hf*16 + 2v + e
__device__ inline v16bf load_a_bf16(const unsigned short* Xrow0, int ld, int k0, int lane) {
  const int m = lane & 15, hf = lane >> 4;
  const unsigned short* p = Xrow0 + (long)m * ld + k0 + (hf << 4);
  frag_bits f;
  f.lo = *(const u32x4*)p;
  f.hi = *(const u32x4*)(p + 8);
  return __builtin_bit_cast(v16bf, f);
}
// B 32x16 bf16: lane n=lane&15, hf=lane>>4; element (2v+e) holds k = hf*16 + 2v + e, column n.
__device__ inline v16bf load_b_bf16(const unsigned short* p) {
  frag_bits f;
  f.lo = *(const u32x4*)p;
  f.hi = *(const u32x4*)(p + 8);
  return __builtin_bit_cast(v16bf, f);
}

// ---- CDNA5 async global->LDS copy: 32 lanes x 16B per instruction, ASYNCcnt ----
__device__ inline void async_cp16(unsigned ldsOff, const unsigned short* gsrc) {
  asm volatile("global_load_async_to_lds_b128 %0, %1, off"
               :: "v"(ldsOff), "v"(gsrc) : "memory");
}
__device__ inline unsigned lds_off(const void* p) {   // generic ptr low 32 bits = LDS byte addr
  return (unsigned)(unsigned long long)p;
}

// =================================================================
// Kernel 0: one-shot fp32 -> bf16 conversion (vectorized 4 elems/thread)
// =================================================================
__global__ void __launch_bounds__(256)
cvt_bf16_kernel(const float* __restrict__ src, unsigned short* __restrict__ dst, int n4) {
  const int i = blockIdx.x * 256 + threadIdx.x;
  if (i >= n4) return;
  f32x4 v = *(const f32x4*)(src + (long)i * 4);
  u16x4 o;
  o.x = f2bf_bits(v.x); o.y = f2bf_bits(v.y);
  o.z = f2bf_bits(v.z); o.w = f2bf_bits(v.w);
  *(u16x4*)(dst + (long)i * 4) = o;
}

// =================================================================
// Kernel 1: kv projection. C[4096,2048] = kv @ Wkv^T (bf16 in, f32 acc).
// fp32 -> d_out kv_proj; bf16 K[S,64] row-major and V^T[64,S] per (b,head).
// =================================================================
__global__ void __launch_bounds__(128)
proj_kv_kernel(const unsigned short* __restrict__ kvb, const unsigned short* __restrict__ Wb,
               float* __restrict__ kvproj, unsigned short* __restrict__ Kbf,
               unsigned short* __restrict__ Vtbf) {
  const int lane = threadIdx.x & 31;
  const int w = threadIdx.x >> 5;
  const int rowBase = blockIdx.x << 4;
  const int colBase = (blockIdx.y << 8) + (w << 6);
  const int n = lane & 15, hf = lane >> 4;
  v8f zero = {};
  v8f acc[4];
#pragma unroll
  for (int t = 0; t < 4; ++t) acc[t] = zero;

  const unsigned short* Xr = kvb + (long)rowBase * kH;
#pragma unroll 2
  for (int k0 = 0; k0 < kH; k0 += 32) {
    __builtin_prefetch(Xr + (long)n * kH + k0 + 64, 0, 3);
    v16bf a = load_a_bf16(Xr, kH, k0, lane);
#pragma unroll
    for (int t = 0; t < 4; ++t) {
      v16bf b = load_b_bf16(Wb + (long)(colBase + (t << 4) + n) * kH + k0 + (hf << 4));
      acc[t] = wmma_bf16(a, b, acc[t]);
    }
  }
#pragma unroll
  for (int t = 0; t < 4; ++t) {
#pragma unroll
    for (int r = 0; r < 8; ++r) {
      const int m = r + (hf << 3);
      const int row = rowBase + m;
      const int col = colBase + (t << 4) + n;
      const float val = acc[t][r];
      kvproj[(long)row * (2 * kH) + col] = val;
      const int b_ = row >> 11, s = row & (kS - 1);
      if (col < kH) {
        const int head = col >> 6, d = col & 63;
        Kbf[((long)(b_ * kNH + head) * kS + s) * kHD + d] = f2bf_bits(val);
      } else {
        const int c2 = col - kH;
        const int head = c2 >> 6, d = c2 & 63;
        Vtbf[((long)(b_ * kNH + head) * kHD + d) * kS + s] = f2bf_bits(val);
      }
    }
  }
}

// =================================================================
// Kernel 2: q projection. query fp32 (residual) + bf16 Q[S,64] per head.
// =================================================================
__global__ void __launch_bounds__(128)
proj_q_kernel(const unsigned short* __restrict__ qb, const unsigned short* __restrict__ Wb,
              float* __restrict__ queryf, unsigned short* __restrict__ Qbf) {
  const int lane = threadIdx.x & 31;
  const int w = threadIdx.x >> 5;
  const int rowBase = blockIdx.x << 4;
  const int colBase = (blockIdx.y << 8) + (w << 6);
  const int n = lane & 15, hf = lane >> 4;
  v8f zero = {};
  v8f acc[4];
#pragma unroll
  for (int t = 0; t < 4; ++t) acc[t] = zero;

  const unsigned short* Xr = qb + (long)rowBase * kH;
#pragma unroll 2
  for (int k0 = 0; k0 < kH; k0 += 32) {
    __builtin_prefetch(Xr + (long)n * kH + k0 + 64, 0, 3);
    v16bf a = load_a_bf16(Xr, kH, k0, lane);
#pragma unroll
    for (int t = 0; t < 4; ++t) {
      v16bf b = load_b_bf16(Wb + (long)(colBase + (t << 4) + n) * kH + k0 + (hf << 4));
      acc[t] = wmma_bf16(a, b, acc[t]);
    }
  }
#pragma unroll
  for (int t = 0; t < 4; ++t) {
#pragma unroll
    for (int r = 0; r < 8; ++r) {
      const int m = r + (hf << 3);
      const int row = rowBase + m;
      const int col = colBase + (t << 4) + n;
      const float val = acc[t][r];
      queryf[(long)row * kH + col] = val;
      const int b_ = row >> 11, s = row & (kS - 1);
      const int head = col >> 6, d = col & 63;
      Qbf[((long)(b_ * kNH + head) * kS + s) * kHD + d] = f2bf_bits(val);
    }
  }
}

// =================================================================
// Kernel 3: flash attention. grid(32 heads, 16 q-blocks), 8 waves x 16 q rows.
// K/V tiles cooperatively staged into LDS with async global->LDS copies
// (double-buffered, ASYNCcnt + barrier), online softmax, P staged as bf16.
// =================================================================
__global__ void __launch_bounds__(256)
attn_kernel(const unsigned short* __restrict__ Qbf, const unsigned short* __restrict__ Kbf,
            const unsigned short* __restrict__ Vtbf, unsigned short* __restrict__ att) {
  __shared__ __align__(16) unsigned short Kt[2][32 * 64];   // [key][d]  4KB/buf
  __shared__ __align__(16) unsigned short Vt[2][64 * 32];   // [d][key]  4KB/buf
  __shared__ __align__(16) unsigned short pst[8][16][32];   // per-wave P tile (bf16)
  const int tid = threadIdx.x;
  const int lane = tid & 31;
  const int w = tid >> 5;
  const int bh = blockIdx.x;                        // b*16 + head
  const int qBlk = blockIdx.y;
  const int qBase = (qBlk << 7) + (w << 4);         // 16 q rows per wave
  const unsigned short* Qh = Qbf + (long)bh * kS * kHD;
  const unsigned short* Kh = Kbf + (long)bh * kS * kHD;
  const unsigned short* Vth = Vtbf + (long)bh * kHD * kS;
  const int n = lane & 15, hf = lane >> 4;

  const v16bf aq0 = load_a_bf16(Qh + (long)qBase * kHD, kHD, 0, lane);
  const v16bf aq1 = load_a_bf16(Qh + (long)qBase * kHD, kHD, 32, lane);

  v8f zero = {};
  v8f acc[4];
#pragma unroll
  for (int t = 0; t < 4; ++t) acc[t] = zero;
  float rmax[8], rsum[8];
#pragma unroll
  for (int r = 0; r < 8; ++r) { rmax[r] = -1e30f; rsum[r] = 0.0f; }

  const int jbMaxW = (qBase + 15) >> 5;             // this wave's causal bound
  const int jbMaxB = (qBlk << 2) + 3;               // block's causal bound

  // cooperative async stage of one 32-key step: 256 threads x 16B for K and V
  auto issue = [&](int jb, int buf) {
    const int j0 = jb << 5;
    async_cp16(lds_off(&Kt[buf][tid * 8]), Kh + (long)j0 * kHD + tid * 8);
    const int row = tid >> 2, c4 = tid & 3;
    async_cp16(lds_off(&Vt[buf][tid * 8]), Vth + (long)row * kS + j0 + c4 * 8);
  };

  issue(0, 0);
  for (int jb = 0; jb <= jbMaxB; ++jb) {
    const int cur = jb & 1;
    asm volatile("s_wait_asynccnt 0x0" ::: "memory");  // own async copies done
    __syncthreads();                                    // everyone's copies done
    if (jb + 1 <= jbMaxB) issue(jb + 1, cur ^ 1);       // prefetch next buffer

    if (jb <= jbMaxW) {
      const int j0 = jb << 5;
      const unsigned short* KtB = &Kt[cur][0];
      const unsigned short* VtB = &Vt[cur][0];
      // ---- scores: 16x16 tiles for keys [j0,j0+15] and [j0+16,j0+31]
      v8f s0 = zero, s1 = zero;
      const unsigned short* pk0 = KtB + n * kHD + (hf << 4);
      const unsigned short* pk1 = KtB + (16 + n) * kHD + (hf << 4);
      s0 = wmma_bf16(aq0, load_b_bf16(pk0), s0);
      s0 = wmma_bf16(aq1, load_b_bf16(pk0 + 32), s0);
      s1 = wmma_bf16(aq0, load_b_bf16(pk1), s1);
      s1 = wmma_bf16(aq1, load_b_bf16(pk1 + 32), s1);

      // ---- scale, causal mask, online softmax (row = r + 8*hf, col = lane&15)
#pragma unroll
      for (int r = 0; r < 8; ++r) {
        const int m = r + (hf << 3);
        const int qRow = qBase + m;
        float v0 = s0[r] * kScale;
        float v1 = s1[r] * kScale;
        if (j0 + n > qRow)      v0 = -60000.0f;
        if (j0 + 16 + n > qRow) v1 = -60000.0f;
        float tmax = fmaxf(v0, v1);
#pragma unroll
        for (int x = 8; x >= 1; x >>= 1) tmax = fmaxf(tmax, __shfl_xor(tmax, x, 32));
        const float nm = fmaxf(rmax[r], tmax);
        const float alpha = __expf(rmax[r] - nm);
        rmax[r] = nm;
        const float p0 = __expf(v0 - nm);
        const float p1 = __expf(v1 - nm);
        float ps = p0 + p1;
#pragma unroll
        for (int x = 8; x >= 1; x >>= 1) ps += __shfl_xor(ps, x, 32);
        rsum[r] = rsum[r] * alpha + ps;
#pragma unroll
        for (int t = 0; t < 4; ++t) acc[t][r] *= alpha;
        pst[w][m][n]      = f2bf_bits(p0);
        pst[w][m][16 + n] = f2bf_bits(p1);
      }
      asm volatile("s_wait_dscnt 0x0" ::: "memory");   // wave-local LDS RAW fence

      // ---- re-fragment P (16x32 bf16) as WMMA A operand: two ds_load_b128
      v16bf ap;
      {
        const unsigned short* pp = &pst[w][lane & 15][hf << 4];
        frag_bits f;
        f.lo = *(const u32x4*)pp;
        f.hi = *(const u32x4*)(pp + 8);
        ap = __builtin_bit_cast(v16bf, f);
      }
      // ---- P @ V from LDS: Vt[d][key] makes B loads contiguous along keys
#pragma unroll
      for (int t = 0; t < 4; ++t) {
        const unsigned short* pv = VtB + ((t << 4) + n) * 32 + (hf << 4);
        acc[t] = wmma_bf16(ap, load_b_bf16(pv), acc[t]);
      }
    }
  }

  // ---- epilogue: attended bf16, row = b*2048+q, col = head*64+d
  const int head = bh & 15, b_ = bh >> 4;
#pragma unroll
  for (int t = 0; t < 4; ++t) {
#pragma unroll
    for (int r = 0; r < 8; ++r) {
      const int m = r + (hf << 3);
      const int row = (b_ << 11) + qBase + m;
      const int col = (head << 6) + (t << 4) + n;
      att[(long)row * kH + col] = f2bf_bits(acc[t][r] / rsum[r]);
    }
  }
}

// =================================================================
// Kernel 4: out projection + residual + LayerNorm. 1 block = 16 rows,
// 8 waves x 128 cols; row buffer in LDS (64 KB), per-row stats via shuffles.
// =================================================================
__global__ void __launch_bounds__(256)
out_ln_kernel(const unsigned short* __restrict__ att, const unsigned short* __restrict__ Wob,
              const float* __restrict__ queryf, const float* __restrict__ g,
              const float* __restrict__ bta, float* __restrict__ out) {
  __shared__ float buf[16 * 1024];
  const int lane = threadIdx.x & 31;
  const int w = threadIdx.x >> 5;
  const int rowBase = blockIdx.x << 4;
  const int n = lane & 15, hf = lane >> 4;
  v8f zero = {};
  v8f acc[8];
#pragma unroll
  for (int t = 0; t < 8; ++t) acc[t] = zero;

  const unsigned short* Xr = att + (long)rowBase * kH;
#pragma unroll 2
  for (int k0 = 0; k0 < kH; k0 += 32) {
    __builtin_prefetch(Xr + (long)n * kH + k0 + 64, 0, 3);
    v16bf a = load_a_bf16(Xr, kH, k0, lane);
#pragma unroll
    for (int t = 0; t < 8; ++t) {
      v16bf b = load_b_bf16(Wob + (long)((w << 7) + (t << 4) + n) * kH + k0 + (hf << 4));
      acc[t] = wmma_bf16(a, b, acc[t]);
    }
  }
#pragma unroll
  for (int t = 0; t < 8; ++t) {
#pragma unroll
    for (int r = 0; r < 8; ++r) {
      const int m = r + (hf << 3);
      const int col = (w << 7) + (t << 4) + n;
      buf[m * 1024 + col] = acc[t][r] + queryf[(long)(rowBase + m) * kH + col];
    }
  }
  __syncthreads();

  const int sub = threadIdx.x & 15;   // 16 threads per row
  const int row = threadIdx.x >> 4;   // 0..15
  float s = 0.0f, s2 = 0.0f;
  for (int c = sub * 64; c < sub * 64 + 64; ++c) {
    const float x = buf[row * 1024 + c];
    s += x;
    s2 += x * x;
  }
#pragma unroll
  for (int x = 8; x >= 1; x >>= 1) {
    s += __shfl_xor(s, x, 32);
    s2 += __shfl_xor(s2, x, 32);
  }
  const float mu = s * (1.0f / 1024.0f);
  const float var = s2 * (1.0f / 1024.0f) - mu * mu;
  const float rstd = rsqrtf(var + kLnEps);
  for (int c = sub * 64; c < sub * 64 + 64; ++c) {
    const float x = buf[row * 1024 + c];
    out[(long)(rowBase + row) * kH + c] = (x - mu) * rstd * g[c] + bta[c];
  }
}

// =================================================================
extern "C" void kernel_launch(void* const* d_in, const int* in_sizes, int n_in,
                              void* d_out, int out_size, void* d_ws, size_t ws_size,
                              hipStream_t stream) {
  (void)in_sizes; (void)n_in; (void)out_size; (void)ws_size;
  const float* q    = (const float*)d_in[0];
  const float* kv   = (const float*)d_in[1];
  /* d_in[2] = bool mask: causal, computed analytically */
  const float* Wkv  = (const float*)d_in[3];
  const float* Wq   = (const float*)d_in[4];
  const float* Wo   = (const float*)d_in[5];
  const float* ln_g = (const float*)d_in[6];
  const float* ln_b = (const float*)d_in[7];

  float* out = (float*)d_out;                              // [4096,1024]
  float* kvproj = out + (size_t)kM * kH;                   // [4096,2048]

  char* ws = (char*)d_ws;
  const size_t MB = 1048576;
  float* queryf         = (float*)ws;                              // 16 MB
  unsigned short* Qbf   = (unsigned short*)(ws + 16 * MB);         //  8 MB
  unsigned short* Kbf   = (unsigned short*)(ws + 24 * MB);         //  8 MB
  unsigned short* Vtbf  = (unsigned short*)(ws + 32 * MB);         //  8 MB
  unsigned short* att   = (unsigned short*)(ws + 40 * MB);         //  8 MB
  unsigned short* kv_bf = (unsigned short*)(ws + 48 * MB);         //  8 MB
  unsigned short* q_bf  = (unsigned short*)(ws + 56 * MB);         //  8 MB
  unsigned short* Wkv_b = (unsigned short*)(ws + 64 * MB);         //  4 MB
  unsigned short* Wq_b  = (unsigned short*)(ws + 68 * MB);         //  2 MB
  unsigned short* Wo_b  = (unsigned short*)(ws + 70 * MB);         //  2 MB

  // one-shot fp32 -> bf16 staging
  auto cvt = [&](const float* s, unsigned short* d, long nelem) {
    const int n4 = (int)(nelem / 4);
    cvt_bf16_kernel<<<(n4 + 255) / 256, 256, 0, stream>>>(s, d, n4);
  };
  cvt(kv,  kv_bf, (long)kM * kH);
  cvt(q,   q_bf,  (long)kM * kH);
  cvt(Wkv, Wkv_b, (long)2 * kH * kH);
  cvt(Wq,  Wq_b,  (long)kH * kH);
  cvt(Wo,  Wo_b,  (long)kH * kH);

  proj_kv_kernel<<<dim3(kM / 16, (2 * kH) / 256), 128, 0, stream>>>(kv_bf, Wkv_b, kvproj, Kbf, Vtbf);
  proj_q_kernel<<<dim3(kM / 16, kH / 256), 128, 0, stream>>>(q_bf, Wq_b, queryf, Qbf);
  attn_kernel<<<dim3(kB * kNH, kS / 128), 256, 0, stream>>>(Qbf, Kbf, Vtbf, att);
  out_ln_kernel<<<dim3(kM / 16), 256, 0, stream>>>(att, Wo_b, queryf, ln_g, ln_b, out);
}
